// LlamaAttentionLSQInteger_74285754351997
// MI455X (gfx1250) — matrix operand order repaired and checked by reading, not compiled, not run on hardware
//
#include <hip/hip_runtime.h>
#include <hip/hip_bf16.h>
#include <math.h>

// ---------------------------------------------------------------------------
// Problem constants (from the reference)
// ---------------------------------------------------------------------------
#define S_LEN 2048
#define HID   2048
#define NH    16
#define NKVH  4
#define HD    128
// LSQ forward: s_eff == |s|  (the g-terms cancel in the forward pass)

typedef __attribute__((ext_vector_type(16))) _Float16 v16h;
typedef __attribute__((ext_vector_type(8)))  _Float16 v8h;
typedef __attribute__((ext_vector_type(8)))  float    v8f;

// ---------------------------------------------------------------------------
// CDNA5 async global->LDS copy (ASYNCcnt-tracked, CDNA5 ISA)
// LDS flat addresses are {aperture, offset[31:0]} (ISA 10.2), so the low 32
// bits of a __shared__ pointer are exactly the LDS byte offset VDST expects.
// ---------------------------------------------------------------------------
__device__ __forceinline__ void async_copy_16B(const void* g, void* l) {
    const unsigned lds_off = (unsigned)(unsigned long long)l;
    asm volatile("global_load_async_to_lds_b128 %0, %1, off"
                 :: "v"(lds_off), "v"(g) : "memory");
}
__device__ __forceinline__ void wait_async0() {
    asm volatile("s_wait_asynccnt 0x0" ::: "memory");
}

// Stage a ROWS x ROWB-byte tile (row stride ldBytes in global) into LDS,
// 16B per lane per async op; ROWS*ROWB must be a multiple of 256*16.
template <int ROWS, int ROWB>
__device__ __forceinline__ void stage_async(const char* __restrict__ g, long ldBytes,
                                            char* l, int tid) {
    constexpr int CPR   = ROWB / 16;          // 16B chunks per row
    constexpr int TOTAL = ROWS * CPR;
#pragma unroll
    for (int i = 0; i < TOTAL / 256; ++i) {
        const int c = tid + i * 256;
        const int r = c / CPR;
        const int o = (c - r * CPR) * 16;
        async_copy_16B(g + (long)r * ldBytes + o, l + r * ROWB + o);
    }
}

// ---------------------------------------------------------------------------
// WMMA fragment loaders from LDS tiles (tile row stride = 32 elements)
// gfx1250 wave32 layouts, CDNA5 ISA 7.12.2:
// A (16x32 f16): lanes 0..15 row M=L, K=0..7 | 16..23; lanes 16..31 K=8..15 | 24..31
// B (32x16 f16): lanes 0..15 col N=L, K=0..7 | 16..23; lanes 16..31 col N=L-16, K=8..15 | 24..31
// ---------------------------------------------------------------------------
__device__ __forceinline__ v16h ldsA_frag_f16(const _Float16* base, int mrel) {
    const int lane = threadIdx.x & 31;
    const int m    = mrel + (lane & 15);
    const int kb   = (lane >> 4) << 3;          // 0 or 8
    const _Float16* p = base + m * 32 + kb;
    union { v16h v; v8h h[2]; } u;
    u.h[0] = *reinterpret_cast<const v8h*>(p);
    u.h[1] = *reinterpret_cast<const v8h*>(p + 16);
    return u.v;
}

__device__ __forceinline__ v16h ldsA_frag_f32(const float* base, int mrel) {
    const int lane = threadIdx.x & 31;
    const int m    = mrel + (lane & 15);
    const int kb   = (lane >> 4) << 3;
    const float* p = base + m * 32 + kb;
    v16h r;
#pragma unroll
    for (int i = 0; i < 8; ++i) {
        r[i]     = (_Float16)p[i];
        r[i + 8] = (_Float16)p[16 + i];
    }
    return r;
}

__device__ __forceinline__ v16h ldsB_frag(const _Float16* base, int nrel) {
    const int lane = threadIdx.x & 31;
    const int n    = nrel + (lane & 15);
    const int kb   = (lane >> 4) << 3;          // 0 or 8
    const _Float16* p = base + n * 32 + kb;
    union { v16h v; v8h h[2]; } u;
    u.h[0] = *reinterpret_cast<const v8h*>(p);
    u.h[1] = *reinterpret_cast<const v8h*>(p + 16);
    return u.v;
}

__device__ __forceinline__ v8f wmma_f16(v16h a, v16h b, v8f c) {
    return __builtin_amdgcn_wmma_f32_16x16x32_f16(
        false, a, false, b, (short)0, c, false, false);
}

// ---------------------------------------------------------------------------
// Generic WMMA GEMM:  C[z] = A[z] * Brm[z/bDiv]^T  (+ epilogue)
// Block = 256 threads = 8 waves (4x2), block tile 128x64, wave tile 32x32.
// A/B tiles are double-buffered in LDS and filled with async global->LDS
// copies; WMMA consumes tile k while the async engine stages tile k+32.
// All problem dims are multiples of the tile sizes (no edge handling).
// ---------------------------------------------------------------------------
enum { EPI_F32 = 0, EPI_SCORES = 1, EPI_LSQ_U8_F16 = 2 };

template <bool A_F32, int EPI>
__global__ __launch_bounds__(256) void gemm_wmma_kernel(
    const void* __restrict__ Av,  int lda, long aBatch,
    const _Float16* __restrict__ B, int ldb, long bBatch, int bDiv,
    void* __restrict__ Cv, int ldc, long cBatch,
    int Kdim, const float* __restrict__ s_ptr)
{
    constexpr int AES = A_F32 ? 4 : 2;                 // A element size (bytes)
    __shared__ __align__(32) char     smemA[2][128 * 32 * AES];
    __shared__ __align__(32) _Float16 smemB[2][64 * 32];

    const int tid  = threadIdx.x;
    const int wave = tid >> 5;
    const int wy   = wave >> 1;                      // 0..3
    const int wx   = wave & 1;                       // 0..1
    const int wm0  = wy * 32;                        // wave row within block tile
    const int wn0  = wx * 32;                        // wave col within block tile
    const int z    = blockIdx.z;

    const long ldaB = (long)lda * AES;
    const long ldbB = (long)ldb * 2;
    const char* Ag = (const char*)Av + (long)z * aBatch * AES
                   + (long)(blockIdx.y * 128) * ldaB;
    const char* Bg = (const char*)B + (long)(z / bDiv) * bBatch * 2
                   + (long)(blockIdx.x * 64) * ldbB;

    v8f acc00 = {}, acc01 = {}, acc10 = {}, acc11 = {};

    // Prologue: stage the first K-slab
    stage_async<128, 32 * AES>(Ag, ldaB, smemA[0], tid);
    stage_async<64, 64>(Bg, ldbB, (char*)smemB[0], tid);
    wait_async0();
    __syncthreads();

    int buf = 0;
    for (int k0 = 0; k0 < Kdim; k0 += 32) {
        const int nbuf = buf ^ 1;
        if (k0 + 32 < Kdim) {   // async-stage next slab while computing this one
            stage_async<128, 32 * AES>(Ag + (long)(k0 + 32) * AES, ldaB, smemA[nbuf], tid);
            stage_async<64, 64>(Bg + (long)(k0 + 32) * 2, ldbB, (char*)smemB[nbuf], tid);
        }

        v16h a0, a1;
        if (A_F32) {
            a0 = ldsA_frag_f32((const float*)smemA[buf], wm0);
            a1 = ldsA_frag_f32((const float*)smemA[buf], wm0 + 16);
        } else {
            a0 = ldsA_frag_f16((const _Float16*)smemA[buf], wm0);
            a1 = ldsA_frag_f16((const _Float16*)smemA[buf], wm0 + 16);
        }
        const v16h b0 = ldsB_frag(smemB[buf], wn0);
        const v16h b1 = ldsB_frag(smemB[buf], wn0 + 16);

        acc00 = wmma_f16(a0, b0, acc00);
        acc01 = wmma_f16(a0, b1, acc01);
        acc10 = wmma_f16(a1, b0, acc10);
        acc11 = wmma_f16(a1, b1, acc11);

        // Producer->consumer handoff for slab k0+32 AND read->overwrite WAR
        // protection for this slab (overwritten at k0+64).
        wait_async0();
        __syncthreads();
        buf = nbuf;
    }

    // C/D f32 16x16 layout: lanes 0..15 -> N=lane, M=r; lanes 16..31 -> N=lane-16, M=8+r
    const int lane = tid & 31;
    const int cn   = lane & 15;
    const int cm   = (lane >> 4) << 3;
    const int m0   = blockIdx.y * 128 + wm0;
    const int n0   = blockIdx.x * 64  + wn0;

    float*    Cf = (float*)Cv    + z * cBatch;
    _Float16* Ch = (_Float16*)Cv + z * cBatch;
    const float sc = (EPI == EPI_LSQ_U8_F16) ? fabsf(s_ptr[0]) : 1.0f;

    const v8f accs[2][2] = { { acc00, acc01 }, { acc10, acc11 } };
#pragma unroll
    for (int i = 0; i < 2; ++i) {
#pragma unroll
        for (int j = 0; j < 2; ++j) {
            const v8f a = accs[i][j];
#pragma unroll
            for (int r = 0; r < 8; ++r) {
                const int row = m0 + i * 16 + cm + r;
                const int col = n0 + j * 16 + cn;
                float v = a[r];
                if (EPI == EPI_SCORES) {
                    v *= 0.08838834764831845f;            // 1/sqrt(128)
                    if (col > row) v = -1.0e9f;           // causal mask
                    Cf[(long)row * ldc + col] = v;
                } else if (EPI == EPI_LSQ_U8_F16) {
                    v = rintf(fminf(fmaxf(v / sc, 0.0f), 255.0f)) * sc;
                    Ch[(long)row * ldc + col] = (_Float16)v;
                } else {
                    Cf[(long)row * ldc + col] = v;
                }
            }
        }
    }
}

// ---------------------------------------------------------------------------
// Elementwise kernels
// ---------------------------------------------------------------------------
__global__ void cvt_f32_f16_kernel(const float* __restrict__ src,
                                   _Float16* __restrict__ dst, long n) {
    const long i = (long)blockIdx.x * blockDim.x + threadIdx.x;
    if (i < n) dst[i] = (_Float16)src[i];
}

// LSQ(sym) + RoPE; in: [S][nheads*HD] f32, out: [nheads][S][HD] f16
__global__ void rope_lsq_kernel(const float* __restrict__ Xf,
                                const float* __restrict__ cosT,
                                const float* __restrict__ sinT,
                                const float* __restrict__ s_ptr,
                                _Float16* __restrict__ out, int nheads) {
    const int width = nheads * HD;
    const long total = (long)S_LEN * width;
    const long idx = (long)blockIdx.x * blockDim.x + threadIdx.x;
    if (idx >= total) return;
    const int s   = (int)(idx / width);
    const int rem = (int)(idx - (long)s * width);
    const int h   = rem / HD;
    const int d   = rem - h * HD;

    const float sc = fabsf(s_ptr[0]);
    float x  = Xf[idx];
    const int d2 = (d < 64) ? d + 64 : d - 64;
    float xp = Xf[(long)s * width + h * HD + d2];
    x  = rintf(fminf(fmaxf(x  / sc, -128.0f), 127.0f)) * sc;
    xp = rintf(fminf(fmaxf(xp / sc, -128.0f), 127.0f)) * sc;
    const float rot = (d < 64) ? -xp : xp;
    const float v = x * cosT[(long)s * HD + d] + rot * sinT[(long)s * HD + d];
    out[((long)h * S_LEN + s) * HD + d] = (_Float16)v;
}

// LSQ(sym) + transpose; in: [S][NKVH*HD] f32, out: Vt [NKVH][HD][S] f16
__global__ void lsq_v_transpose_kernel(const float* __restrict__ Vf,
                                       const float* __restrict__ s_ptr,
                                       _Float16* __restrict__ Vt) {
    const int width = NKVH * HD;
    const long total = (long)S_LEN * width;
    const long idx = (long)blockIdx.x * blockDim.x + threadIdx.x;
    if (idx >= total) return;
    const int s   = (int)(idx / width);
    const int rem = (int)(idx - (long)s * width);
    const int kv  = rem / HD;
    const int d   = rem - kv * HD;
    const float sc = fabsf(s_ptr[0]);
    const float v = rintf(fminf(fmaxf(Vf[idx] / sc, -128.0f), 127.0f)) * sc;
    Vt[((long)kv * HD + d) * S_LEN + s] = (_Float16)v;
}

// Row softmax + unsigned LSQ, in place on the attn region of d_out.
__global__ __launch_bounds__(256) void softmax_lsq_kernel(float* __restrict__ attn,
                                                          const float* __restrict__ s_ptr) {
    __shared__ float red[256];
    const long row = blockIdx.x;
    const long h   = blockIdx.y;
    float* p = attn + (h * S_LEN + row) * S_LEN;
    const int t = threadIdx.x;

    float vals[8];
    float lmax = -3.4e38f;
#pragma unroll
    for (int i = 0; i < 8; ++i) { vals[i] = p[i * 256 + t]; lmax = fmaxf(lmax, vals[i]); }
    red[t] = lmax; __syncthreads();
    for (int o = 128; o > 0; o >>= 1) { if (t < o) red[t] = fmaxf(red[t], red[t + o]); __syncthreads(); }
    const float m = red[0]; __syncthreads();

    float lsum = 0.0f;
#pragma unroll
    for (int i = 0; i < 8; ++i) { vals[i] = __expf(vals[i] - m); lsum += vals[i]; }
    red[t] = lsum; __syncthreads();
    for (int o = 128; o > 0; o >>= 1) { if (t < o) red[t] += red[t + o]; __syncthreads(); }
    const float inv = 1.0f / red[0];

    const float sc = fabsf(s_ptr[0]);
#pragma unroll
    for (int i = 0; i < 8; ++i) {
        float q = vals[i] * inv;
        q = rintf(fminf(fmaxf(q / sc, 0.0f), 255.0f)) * sc;
        p[i * 256 + t] = q;
    }
}

// ---------------------------------------------------------------------------
// Host orchestration
// ---------------------------------------------------------------------------
extern "C" void kernel_launch(void* const* d_in, const int* in_sizes, int n_in,
                              void* d_out, int out_size, void* d_ws, size_t ws_size,
                              hipStream_t stream) {
    const float* hidden = (const float*)d_in[0];
    const float* Wq     = (const float*)d_in[1];
    const float* Wk     = (const float*)d_in[2];
    const float* Wv     = (const float*)d_in[3];
    const float* Wo     = (const float*)d_in[4];
    const float* s_q    = (const float*)d_in[5];
    const float* s_k    = (const float*)d_in[6];
    const float* s_v    = (const float*)d_in[7];
    const float* s_attn = (const float*)d_in[8];
    const float* s_out  = (const float*)d_in[9];
    const float* cosT   = (const float*)d_in[10];
    const float* sinT   = (const float*)d_in[11];
    // d_in[12] attention_mask: causal mask applied analytically in the scores epilogue

    float* outF  = (float*)d_out;                         // [S][HID]
    float* attnF = (float*)d_out + (long)S_LEN * HID;     // [NH][S][S]

    // Workspace layout (all 256B-aligned by construction)
    char* ws = (char*)d_ws;
    _Float16* Xh   = (_Float16*)(ws);                       // 8 MB  [S][HID]
    _Float16* Wqh  = (_Float16*)(ws + 8388608);             // 8 MB  [HID][HID]
    _Float16* Wkh  = (_Float16*)(ws + 16777216);            // 2 MB  [512][HID]
    _Float16* Wvh  = (_Float16*)(ws + 18874368);            // 2 MB  [512][HID]
    _Float16* Woh  = (_Float16*)(ws + 20971520);            // 8 MB  [HID][HID]
    float*    Qf32 = (float*)   (ws + 29360128);            // 16 MB [S][HID]
    float*    Kf32 = (float*)   (ws + 46137344);            // 4 MB  [S][512]
    float*    Vf32 = (float*)   (ws + 50331648);            // 4 MB  [S][512]
    _Float16* Qh   = (_Float16*)(ws + 54525952);            // 8 MB  [NH][S][HD]
    _Float16* Kh   = (_Float16*)(ws + 62914560);            // 2 MB  [NKVH][S][HD]
    _Float16* Vt   = (_Float16*)(ws + 65011712);            // 2 MB  [NKVH][HD][S]
    _Float16* Oh   = (_Float16*)(ws + 67108864);            // 8 MB  [S][NH*HD]

    const dim3 blk(256);

    // 1) f32 -> f16 conversions
    cvt_f32_f16_kernel<<<(4194304 + 255) / 256, blk, 0, stream>>>(hidden, Xh, 4194304L);
    cvt_f32_f16_kernel<<<(4194304 + 255) / 256, blk, 0, stream>>>(Wq, Wqh, 4194304L);
    cvt_f32_f16_kernel<<<(1048576 + 255) / 256, blk, 0, stream>>>(Wk, Wkh, 1048576L);
    cvt_f32_f16_kernel<<<(1048576 + 255) / 256, blk, 0, stream>>>(Wv, Wvh, 1048576L);
    cvt_f32_f16_kernel<<<(4194304 + 255) / 256, blk, 0, stream>>>(Wo, Woh, 4194304L);

    // 2) QKV projections: X[S][HID] @ W^T -> f32
    gemm_wmma_kernel<false, EPI_F32><<<dim3(HID / 64, S_LEN / 128, 1), blk, 0, stream>>>(
        Xh, HID, 0L, Wqh, HID, 0L, 1, Qf32, HID, 0L, HID, nullptr);
    gemm_wmma_kernel<false, EPI_F32><<<dim3(512 / 64, S_LEN / 128, 1), blk, 0, stream>>>(
        Xh, HID, 0L, Wkh, HID, 0L, 1, Kf32, 512, 0L, HID, nullptr);
    gemm_wmma_kernel<false, EPI_F32><<<dim3(512 / 64, S_LEN / 128, 1), blk, 0, stream>>>(
        Xh, HID, 0L, Wvh, HID, 0L, 1, Vf32, 512, 0L, HID, nullptr);

    // 3) LSQ + RoPE (Q, K); LSQ + transpose (V)
    rope_lsq_kernel<<<(4194304 + 255) / 256, blk, 0, stream>>>(Qf32, cosT, sinT, s_q, Qh, NH);
    rope_lsq_kernel<<<(1048576 + 255) / 256, blk, 0, stream>>>(Kf32, cosT, sinT, s_k, Kh, NKVH);
    lsq_v_transpose_kernel<<<(1048576 + 255) / 256, blk, 0, stream>>>(Vf32, s_v, Vt);

    // 4) scores = Q_h @ K_h^T / sqrt(HD) + causal mask  (GQA: kv head = h/4)
    gemm_wmma_kernel<false, EPI_SCORES><<<dim3(S_LEN / 64, S_LEN / 128, NH), blk, 0, stream>>>(
        Qh, HD, (long)S_LEN * HD, Kh, HD, (long)S_LEN * HD, NH / NKVH,
        attnF, S_LEN, (long)S_LEN * S_LEN, HD, nullptr);

    // 5) softmax + LSQ(u8), in place -> final attn output
    softmax_lsq_kernel<<<dim3(S_LEN, NH), blk, 0, stream>>>(attnF, s_attn);

    // 6) O_h = attn_h @ V_h (A is f32 attn, converted per-fragment), LSQ(u8) -> f16
    gemm_wmma_kernel<true, EPI_LSQ_U8_F16><<<dim3(HD / 64, S_LEN / 128, NH), blk, 0, stream>>>(
        attnF, S_LEN, (long)S_LEN * S_LEN, Vt, S_LEN, (long)HD * S_LEN, NH / NKVH,
        Oh, HID, (long)HD, S_LEN, s_out);

    // 7) out = O @ Wo^T -> f32
    gemm_wmma_kernel<false, EPI_F32><<<dim3(HID / 64, S_LEN / 128, 1), blk, 0, stream>>>(
        Oh, HID, 0L, Woh, HID, 0L, 1, outF, HID, 0L, HID, nullptr);
}
